// SAIGA_44495861187340
// MI455X (gfx1250) — matrix operand order repaired and checked by reading, not compiled, not run on hardware
//
#include <hip/hip_runtime.h>

typedef _Float16 v16h __attribute__((ext_vector_type(16)));
typedef _Float16 v8h  __attribute__((ext_vector_type(8)));
typedef float    v8f  __attribute__((ext_vector_type(8)));

#define B_  4
#define S_  2048
#define D_  512
#define H_  4
#define HD_ 128
#define ROWS (B_ * S_)          // 8192
#define DEXP (2 * D_)           // 1024

__device__ __forceinline__ v8f wmma16(v16h a, v16h b, v8f c) {
    return __builtin_amdgcn_wmma_f32_16x16x32_f16(
        /*neg_a=*/false, a, /*neg_b=*/false, b,
        /*c_mod=*/(short)0, c, /*reuse_a=*/false, /*reuse_b=*/false);
}

__device__ __forceinline__ v8f zero8() {
    v8f z = {0.f, 0.f, 0.f, 0.f, 0.f, 0.f, 0.f, 0.f};
    return z;
}

// Build A-fragment (16x32 f16): lane row = lane&15; K = half*8 + e (+8 for e>=8)
__device__ __forceinline__ v16h load_a_frag(const _Float16* rowbase, int half) {
    v16h t;
    ((v8h*)&t)[0] = *(const v8h*)(rowbase + half * 8);
    ((v8h*)&t)[1] = *(const v8h*)(rowbase + 16 + half * 8);
    return t;
}
// Build B-fragment (32x16 f16): lane col = lane&15; K = half*8 + e (+8 for e>=8)
__device__ __forceinline__ v16h load_b_frag(const _Float16* colbase, int half) {
    v16h t;
    ((v8h*)&t)[0] = *(const v8h*)(colbase + half * 8);
    ((v8h*)&t)[1] = *(const v8h*)(colbase + 16 + half * 8);
    return t;
}

// ---------------- elementwise f32 -> f16 cast ----------------
__global__ void cast_f16_kernel(const float* __restrict__ x, _Float16* __restrict__ y, int n) {
    int i = blockIdx.x * blockDim.x + threadIdx.x;
    if (i < n) y[i] = (_Float16)x[i];
}

// ---------------- W[K][N] f32  ->  Wt[N][K] f16 ----------------
__global__ void transpose_cast_kernel(const float* __restrict__ W, _Float16* __restrict__ Wt,
                                      int Krows, int Ncols) {
    int idx = blockIdx.x * blockDim.x + threadIdx.x;
    if (idx >= Krows * Ncols) return;
    int k = idx / Ncols, n = idx - k * Ncols;
    Wt[(size_t)n * Krows + k] = (_Float16)W[idx];
}

// ---------------- tiled WMMA GEMM (double-buffered LDS) ----------------
// C[M][N] = A[M][K] (f16) * Bt[N][K]^T (f16) + bias
// EPI: 0 = relu -> f16 ; 1 = +resid -> f32 ; 2 = -> f16 ; 3 = -> f16, per-head transposed (Vt)
template <int EPI>
__global__ __launch_bounds__(256) void gemm_wmma(
    const _Float16* __restrict__ A, const _Float16* __restrict__ Bt,
    const float* __restrict__ bias, const float* __restrict__ resid,
    float* __restrict__ outF, _Float16* __restrict__ outH,
    int M, int N, int K) {
    __shared__ __align__(16) _Float16 lds_a[2][128][32];
    __shared__ __align__(16) _Float16 lds_b[2][128][32];

    const int tid  = threadIdx.x;
    const int lane = tid & 31, wid = tid >> 5;
    const int half = lane >> 4, ncol = lane & 15;
    const int wm = wid >> 1, wn = wid & 1;          // 4x2 wave grid
    const int m0 = blockIdx.y * 128, n0 = blockIdx.x * 128;

    v8f acc[2][4];
    for (int mi = 0; mi < 2; mi++)
        for (int ni = 0; ni < 4; ni++) acc[mi][ni] = zero8();

    const int ldrow = tid >> 1, ldseg = tid & 1;    // 256 threads fill 128x32 tiles
    const _Float16* gA = A + (size_t)(m0 + ldrow) * K + ldseg * 16;
    const _Float16* gB = Bt + (size_t)(n0 + ldrow) * K + ldseg * 16;

    const int nk = K >> 5;
    // stage tile 0 into buffer 0
    {
        const v8h* sa = (const v8h*)gA;
        v8h* da = (v8h*)&lds_a[0][ldrow][ldseg * 16];
        da[0] = sa[0]; da[1] = sa[1];
        const v8h* sb = (const v8h*)gB;
        v8h* db = (v8h*)&lds_b[0][ldrow][ldseg * 16];
        db[0] = sb[0]; db[1] = sb[1];
    }

    for (int t = 0; t < nk; ++t) {
        __syncthreads();
        const int cur = t & 1;
        if (t + 1 < nk) {       // fill the other buffer while computing on this one
            const v8h* sa = (const v8h*)(gA + (t + 1) * 32);
            v8h* da = (v8h*)&lds_a[cur ^ 1][ldrow][ldseg * 16];
            da[0] = sa[0]; da[1] = sa[1];
            const v8h* sb = (const v8h*)(gB + (t + 1) * 32);
            v8h* db = (v8h*)&lds_b[cur ^ 1][ldrow][ldseg * 16];
            db[0] = sb[0]; db[1] = sb[1];
        }
        if (t + 2 < nk) {       // L2 prefetch two tiles ahead
            __builtin_prefetch(gA + (t + 2) * 32, 0, 1);
            __builtin_prefetch(gB + (t + 2) * 32, 0, 1);
        }

        v16h af[2], bf[4];
#pragma unroll
        for (int mi = 0; mi < 2; mi++)
            af[mi] = load_a_frag(&lds_a[cur][wm * 32 + mi * 16 + ncol][0], half);
#pragma unroll
        for (int ni = 0; ni < 4; ni++)
            bf[ni] = load_b_frag(&lds_b[cur][wn * 64 + ni * 16 + ncol][0], half);
#pragma unroll
        for (int ni = 0; ni < 4; ni++)       // ni-major: consecutive WMMAs independent
#pragma unroll
            for (int mi = 0; mi < 2; mi++)
                acc[mi][ni] = wmma16(af[mi], bf[ni], acc[mi][ni]);
    }

#pragma unroll
    for (int mi = 0; mi < 2; mi++)
#pragma unroll
        for (int ni = 0; ni < 4; ni++) {
            int gc = n0 + wn * 64 + ni * 16 + ncol;
            float bia = bias[gc];
#pragma unroll
            for (int r = 0; r < 8; r++) {
                int gr = m0 + wm * 32 + mi * 16 + half * 8 + r;
                float v = acc[mi][ni][r] + bia;
                if (EPI == 0) {
                    outH[(size_t)gr * N + gc] = (_Float16)fmaxf(v, 0.f);
                } else if (EPI == 1) {
                    outF[(size_t)gr * N + gc] = v + resid[(size_t)gr * N + gc];
                } else if (EPI == 2) {
                    outH[(size_t)gr * N + gc] = (_Float16)v;
                } else {
                    int bb = gr >> 11, ss = gr & 2047, hh = gc >> 7, dl = gc & 127;
                    outH[((size_t)(bb * H_ + hh) * HD_ + dl) * S_ + ss] = (_Float16)v;
                }
            }
        }
}

// ---------------- LayerNorm: one wave per 512-wide row ----------------
template <bool ADD, bool STOREH>
__global__ __launch_bounds__(256) void ln_rows(
    const float* __restrict__ X, const float* __restrict__ Y,
    const float* __restrict__ gamma, const float* __restrict__ beta,
    float* __restrict__ outF, _Float16* __restrict__ outH, int nrows) {
    int wid = threadIdx.x >> 5, lane = threadIdx.x & 31;
    int row = blockIdx.x * 8 + wid;
    if (row >= nrows) return;
    const float* xp = X + (size_t)row * D_ + lane * 16;
    const float* yp = ADD ? (Y + (size_t)row * D_ + lane * 16) : nullptr;
    float v[16];
    float s = 0.f, ss = 0.f;
#pragma unroll
    for (int u = 0; u < 16; u++) {
        float t = xp[u];
        if (ADD) t += yp[u];
        v[u] = t; s += t; ss += t * t;
    }
#pragma unroll
    for (int m = 1; m < 32; m <<= 1) {
        s  += __shfl_xor(s, m, 32);
        ss += __shfl_xor(ss, m, 32);
    }
    float mu = s * (1.f / D_);
    float var = ss * (1.f / D_) - mu * mu;
    float rs = rsqrtf(var + 1e-5f);
    const float* gp = gamma + lane * 16;
    const float* bp = beta + lane * 16;
#pragma unroll
    for (int u = 0; u < 16; u++) {
        float o = (v[u] - mu) * rs * gp[u] + bp[u];
        outF[(size_t)row * D_ + lane * 16 + u] = o;
        if (STOREH) outH[(size_t)row * D_ + lane * 16 + u] = (_Float16)o;
    }
}

// ---------------- flash attention with distance bias ----------------
// grid: B*H*(S/64) blocks, 128 threads (4 waves); wave = 16 query rows.
__global__ __launch_bounds__(128) void attn_kernel(
    const _Float16* __restrict__ Q,   // [ROWS][D]
    const _Float16* __restrict__ Km,  // [ROWS][D]
    const _Float16* __restrict__ Vt,  // [B*H*HD][S]
    const float* __restrict__ pos,    // [B][S][3]
    const float* __restrict__ alpha_p,
    float* __restrict__ ctx) {        // [ROWS][D]
    const int nqt = S_ / 64;
    int qt = blockIdx.x % nqt;
    int hh = (blockIdx.x / nqt) & (H_ - 1);
    int b  = blockIdx.x / (nqt * H_);
    int tid = threadIdx.x, wid = tid >> 5, lane = tid & 31;
    int half = lane >> 4, ncol = lane & 15;
    int qbase = qt * 64 + wid * 16;
    size_t tok0 = (size_t)b * S_;
    float alpha = alpha_p[0];
    const float scale = 0.08838834764831845f;  // 1/sqrt(128)

    __shared__ __align__(16) _Float16 lds_p[4][16][32];

    // Q fragments (kept in registers, reused across all key tiles)
    v16h qf[4];
    {
        const _Float16* qp = Q + (tok0 + qbase + ncol) * D_ + hh * HD_;
#pragma unroll
        for (int kc = 0; kc < 4; kc++) qf[kc] = load_a_frag(qp + kc * 32, half);
    }
    // query positions for this lane's 8 C-rows
    float px[8], py[8], pz[8];
#pragma unroll
    for (int r = 0; r < 8; r++) {
        int i = qbase + half * 8 + r;
        const float* pp = pos + (tok0 + i) * 3;
        px[r] = pp[0]; py[r] = pp[1]; pz[r] = pp[2];
    }

    float mrow[8], lrow[8];
    v8f acc[8];
#pragma unroll
    for (int r = 0; r < 8; r++) { mrow[r] = -__builtin_inff(); lrow[r] = 0.f; }
#pragma unroll
    for (int f = 0; f < 8; f++) acc[f] = zero8();

    const _Float16* vbase = Vt + (size_t)(b * H_ + hh) * HD_ * S_;

    for (int j0 = 0; j0 < S_; j0 += 32) {
        // L2 prefetch of the next key tile (K rows + Vt rows)
        if (j0 + 32 < S_) {
            const _Float16* kp_n = Km + (tok0 + j0 + 32 + lane) * D_ + hh * HD_;
            __builtin_prefetch(kp_n, 0, 1);
            __builtin_prefetch(kp_n + 64, 0, 1);
#pragma unroll
            for (int u = 0; u < 4; u++)
                __builtin_prefetch(vbase + (size_t)(lane + u * 32) * S_ + j0 + 32, 0, 1);
        }
        // ---- scores: Q (16x128) x K^T tile (128x32), two independent chains ----
        const _Float16* kp0 = Km + (tok0 + j0 + ncol) * D_ + hh * HD_;
        const _Float16* kp1 = kp0 + 16 * D_;
        v8f a0 = zero8(), a1 = zero8();
#pragma unroll
        for (int kc = 0; kc < 4; kc++) {
            v16h b0 = load_b_frag(kp0 + kc * 32, half);
            v16h b1 = load_b_frag(kp1 + kc * 32, half);
            a0 = wmma16(qf[kc], b0, a0);
            a1 = wmma16(qf[kc], b1, a1);
        }
        v8f sc[2] = {a0, a1};
        float kx[2], ky[2], kz[2];
#pragma unroll
        for (int nc = 0; nc < 2; nc++) {
            const float* pp = pos + (tok0 + j0 + nc * 16 + ncol) * 3;
            kx[nc] = pp[0]; ky[nc] = pp[1]; kz[nc] = pp[2];
        }
        // ---- scale + distance bias + local row max ----
        float mloc[8];
#pragma unroll
        for (int r = 0; r < 8; r++) {
            float dx0 = px[r] - kx[0], dy0 = py[r] - ky[0], dz0 = pz[r] - kz[0];
            float dx1 = px[r] - kx[1], dy1 = py[r] - ky[1], dz1 = pz[r] - kz[1];
            float s0 = sc[0][r] * scale - alpha * (dx0 * dx0 + dy0 * dy0 + dz0 * dz0);
            float s1 = sc[1][r] * scale - alpha * (dx1 * dx1 + dy1 * dy1 + dz1 * dz1);
            sc[0][r] = s0; sc[1][r] = s1;
            mloc[r] = fmaxf(s0, s1);
        }
#pragma unroll
        for (int m = 1; m < 16; m <<= 1)
#pragma unroll
            for (int r = 0; r < 8; r++) mloc[r] = fmaxf(mloc[r], __shfl_xor(mloc[r], m, 32));
        // ---- online softmax update ----
        float pr[2][8], rsum[8];
#pragma unroll
        for (int r = 0; r < 8; r++) {
            float mn = fmaxf(mrow[r], mloc[r]);
            float rescale = __expf(mrow[r] - mn);
            mrow[r] = mn;
            lrow[r] *= rescale;
#pragma unroll
            for (int f = 0; f < 8; f++) acc[f][r] *= rescale;
            float p0 = __expf(sc[0][r] - mn), p1 = __expf(sc[1][r] - mn);
            pr[0][r] = p0; pr[1][r] = p1;
            rsum[r] = p0 + p1;
        }
#pragma unroll
        for (int m = 1; m < 16; m <<= 1)
#pragma unroll
            for (int r = 0; r < 8; r++) rsum[r] += __shfl_xor(rsum[r], m, 32);
#pragma unroll
        for (int r = 0; r < 8; r++) lrow[r] += rsum[r];
        // ---- transpose P from C-layout to A-layout via LDS ----
#pragma unroll
        for (int nc = 0; nc < 2; nc++)
#pragma unroll
            for (int r = 0; r < 8; r++)
                lds_p[wid][half * 8 + r][nc * 16 + ncol] = (_Float16)pr[nc][r];
        __syncthreads();
        v16h pf = load_a_frag(&lds_p[wid][ncol][0], half);
        __syncthreads();
        // ---- ctx += P (16x32) x V tile (32x128): 8 independent accumulators ----
        const _Float16* vb = vbase + j0;
#pragma unroll
        for (int f = 0; f < 8; f++) {
            int d = f * 16 + ncol;
            acc[f] = wmma16(pf, load_b_frag(vb + (size_t)d * S_, half), acc[f]);
        }
    }
    // ---- normalize and store ctx ----
#pragma unroll
    for (int f = 0; f < 8; f++)
#pragma unroll
        for (int r = 0; r < 8; r++) {
            int i = qbase + half * 8 + r;
            int c = hh * HD_ + f * 16 + ncol;
            ctx[(tok0 + i) * D_ + c] = acc[f][r] / lrow[r];
        }
}

extern "C" void kernel_launch(void* const* d_in, const int* in_sizes, int n_in,
                              void* d_out, int out_size, void* d_ws, size_t ws_size,
                              hipStream_t stream) {
    const float* X     = (const float*)d_in[0];
    const float* pos   = (const float*)d_in[1];
    const float* Wexp  = (const float*)d_in[2];
    const float* bexp  = (const float*)d_in[3];
    const float* Wsq   = (const float*)d_in[4];
    const float* bsq   = (const float*)d_in[5];
    const float* gamma = (const float*)d_in[6];
    const float* beta  = (const float*)d_in[7];
    const float* Wq    = (const float*)d_in[8];
    const float* bq    = (const float*)d_in[9];
    const float* Wk    = (const float*)d_in[10];
    const float* bk    = (const float*)d_in[11];
    const float* Wv    = (const float*)d_in[12];
    const float* bv    = (const float*)d_in[13];
    const float* alpha = (const float*)d_in[14];
    float* out = (float*)d_out;

    char* ws = (char*)d_ws;
    size_t off = 0;
    auto alloc = [&](size_t bytes) -> char* {
        char* p = ws + off;
        off = (off + bytes + 255) & ~(size_t)255;
        return p;
    };
    _Float16* xh     = (_Float16*)alloc((size_t)ROWS * D_ * 2);
    _Float16* Wexp_t = (_Float16*)alloc((size_t)DEXP * D_ * 2);
    _Float16* Wsq_t  = (_Float16*)alloc((size_t)D_ * DEXP * 2);
    _Float16* Wq_t   = (_Float16*)alloc((size_t)D_ * D_ * 2);
    _Float16* Wk_t   = (_Float16*)alloc((size_t)D_ * D_ * 2);
    _Float16* Wv_t   = (_Float16*)alloc((size_t)D_ * D_ * 2);
    _Float16* hbuf   = (_Float16*)alloc((size_t)ROWS * DEXP * 2);
    float*    se     = (float*)alloc((size_t)ROWS * D_ * 4);
    _Float16* seh    = (_Float16*)alloc((size_t)ROWS * D_ * 2);
    _Float16* qh     = (_Float16*)alloc((size_t)ROWS * D_ * 2);
    _Float16* kh     = (_Float16*)alloc((size_t)ROWS * D_ * 2);
    _Float16* vt     = (_Float16*)alloc((size_t)ROWS * D_ * 2);
    float*    ctx    = (float*)alloc((size_t)ROWS * D_ * 4);

    // 0) precision conversions / weight transposes
    cast_f16_kernel<<<(ROWS * D_ + 255) / 256, 256, 0, stream>>>(X, xh, ROWS * D_);
    transpose_cast_kernel<<<(D_ * DEXP + 255) / 256, 256, 0, stream>>>(Wexp, Wexp_t, D_, DEXP);
    transpose_cast_kernel<<<(DEXP * D_ + 255) / 256, 256, 0, stream>>>(Wsq, Wsq_t, DEXP, D_);
    transpose_cast_kernel<<<(D_ * D_ + 255) / 256, 256, 0, stream>>>(Wq, Wq_t, D_, D_);
    transpose_cast_kernel<<<(D_ * D_ + 255) / 256, 256, 0, stream>>>(Wk, Wk_t, D_, D_);
    transpose_cast_kernel<<<(D_ * D_ + 255) / 256, 256, 0, stream>>>(Wv, Wv_t, D_, D_);

    // 1) FFN expand + relu -> h (f16)
    gemm_wmma<0><<<dim3(DEXP / 128, ROWS / 128), 256, 0, stream>>>(
        xh, Wexp_t, bexp, nullptr, nullptr, hbuf, ROWS, DEXP, D_);
    // 2) FFN squeeze + bias + residual -> se (pre-LN, f32)
    gemm_wmma<1><<<dim3(D_ / 128, ROWS / 128), 256, 0, stream>>>(
        hbuf, Wsq_t, bsq, X, se, nullptr, ROWS, D_, DEXP);
    // 3) LayerNorm (in place) -> se f32 + f16
    ln_rows<false, true><<<ROWS / 8, 256, 0, stream>>>(se, nullptr, gamma, beta, se, seh, ROWS);
    // 4) Q/K/V projections
    gemm_wmma<2><<<dim3(D_ / 128, ROWS / 128), 256, 0, stream>>>(
        seh, Wq_t, bq, nullptr, nullptr, qh, ROWS, D_, D_);
    gemm_wmma<2><<<dim3(D_ / 128, ROWS / 128), 256, 0, stream>>>(
        seh, Wk_t, bk, nullptr, nullptr, kh, ROWS, D_, D_);
    gemm_wmma<3><<<dim3(D_ / 128, ROWS / 128), 256, 0, stream>>>(
        seh, Wv_t, bv, nullptr, nullptr, vt, ROWS, D_, D_);
    // 5) attention with distance bias -> ctx (f32)
    attn_kernel<<<B_ * H_ * (S_ / 64), 128, 0, stream>>>(qh, kh, vt, pos, alpha, ctx);
    // 6) final LayerNorm(se + ctx) -> out
    ln_rows<true, false><<<ROWS / 8, 256, 0, stream>>>(se, ctx, gamma, beta, out, nullptr, ROWS);
}
